// Net_40321152975322
// MI455X (gfx1250) — matrix operand order repaired and checked by reading, not compiled, not run on hardware
//
#include <hip/hip_runtime.h>
#include <hip/hip_bf16.h>

typedef __attribute__((ext_vector_type(16))) _Float16 v16h;
typedef __attribute__((ext_vector_type(4)))  _Float16 h4;
typedef __attribute__((ext_vector_type(8)))  float    v8f;

#define TM 64
#define TN 64
#define TK 32
#define LDP 40   // padded LDS row length in halves: 80 B stride (multiple of 16 B)

// ---------------------------------------------------------------------------
// Farthest point sampling: one block per cloud.
// Positions staged in LDS once; mind[] lives in registers (j = tid + 256*t);
// per-step argmax uses wave32 shuffle reduction -> only 2 barriers per step.
// ---------------------------------------------------------------------------
__global__ __launch_bounds__(256) void fps_kernel(const float* __restrict__ pos,
                                                  int npts, int nsample,
                                                  int* __restrict__ idx)
{
    __shared__ float ps[4096 * 3];
    __shared__ float redv[8];
    __shared__ int   redi[8];
    __shared__ int   s_last;

    const int b    = blockIdx.x;
    const int tid  = threadIdx.x;
    const int lane = tid & 31;
    const int wv   = tid >> 5;
    const float* p = pos + (size_t)b * npts * 3;

    for (int i = tid; i < npts * 3; i += 256) ps[i] = p[i];

    float mind[16];
#pragma unroll
    for (int t = 0; t < 16; ++t) mind[t] = 1e10f;

    if (tid == 0) { idx[(size_t)b * nsample] = b * npts; s_last = 0; }
    __syncthreads();

    for (int s = 1; s < nsample; ++s) {
        const int last = s_last;
        const float lx = ps[last*3], ly = ps[last*3+1], lz = ps[last*3+2];
        float bestv = -1.0f; int besti = 0;
#pragma unroll 4
        for (int t = 0; t < 16; ++t) {
            const int j = tid + (t << 8);
            if (j < npts) {
                const float dx = ps[j*3]-lx, dy = ps[j*3+1]-ly, dz = ps[j*3+2]-lz;
                const float d  = dx*dx + dy*dy + dz*dz;
                const float m  = fminf(mind[t], d);
                mind[t] = m;
                if (m > bestv) { bestv = m; besti = j; }
            }
        }
        // wave32 shuffle argmax (larger value wins; tie -> smaller index)
#pragma unroll
        for (int o = 16; o > 0; o >>= 1) {
            const float ov = __shfl_xor(bestv, o);
            const int   oi = __shfl_xor(besti, o);
            if (ov > bestv || (ov == bestv && oi < besti)) { bestv = ov; besti = oi; }
        }
        if (lane == 0) { redv[wv] = bestv; redi[wv] = besti; }
        __syncthreads();
        if (tid == 0) {
            float bv = redv[0]; int bi = redi[0];
            for (int w2 = 1; w2 < 8; ++w2)
                if (redv[w2] > bv || (redv[w2] == bv && redi[w2] < bi)) { bv = redv[w2]; bi = redi[w2]; }
            s_last = bi;
            idx[(size_t)b * nsample + s] = b * npts + bi;
        }
        __syncthreads();
    }
}

// ---------------------------------------------------------------------------
__global__ void gather3(const float* __restrict__ src, const int* __restrict__ idx,
                        float* __restrict__ dst, int M)
{
    int i = blockIdx.x * blockDim.x + threadIdx.x;
    if (i >= M * 3) return;
    int r = i / 3, c = i % 3;
    dst[i] = src[(size_t)idx[r] * 3 + c];
}

// ---------------------------------------------------------------------------
// Ball query: one wave32 per center; ballot compaction of in-radius points.
// ---------------------------------------------------------------------------
__global__ void ball_query(const float* __restrict__ pts,
                           const float* __restrict__ centers, float r2,
                           int* __restrict__ nbr, int* __restrict__ cnt,
                           int Bn, int Ns, int M)
{
    const int gwave = (blockIdx.x * blockDim.x + threadIdx.x) >> 5;
    const int lane  = threadIdx.x & 31;
    const int total = Bn * M;
    if (gwave >= total) return;
    const int b = gwave / M;
    const float* cp = centers + (size_t)gwave * 3;
    const float qx = cp[0], qy = cp[1], qz = cp[2];
    const float* pb = pts + (size_t)b * Ns * 3;

    int count = 0;
    for (int base = 0; base < Ns && count < 64; base += 32) {
        const int j = base + lane;
        bool ok = false;
        if (j < Ns) {
            const float dx = pb[j*3]-qx, dy = pb[j*3+1]-qy, dz = pb[j*3+2]-qz;
            ok = (dx*dx + dy*dy + dz*dz) <= r2;
        }
        unsigned long long m = __ballot(ok);
        int pre = __popcll(m & ((1ull << lane) - 1ull));
        if (ok && (count + pre) < 64)
            nbr[(size_t)gwave * 64 + count + pre] = b * Ns + j;
        count += __popcll(m);
    }
    if (lane == 0) cnt[gwave] = count < 64 ? count : 64;
}

// ---------------------------------------------------------------------------
// Build PointNetConv features with padded row pitch Cpad (zero-filled pad).
// ---------------------------------------------------------------------------
__global__ void build_sa_feat(const float* __restrict__ xsrc, int Cx,
                              const float* __restrict__ psrc,
                              const float* __restrict__ q,
                              const int* __restrict__ nbr,
                              const int* __restrict__ cnt,
                              float* __restrict__ F, int Mtot, int Cpad)
{
    int row = blockIdx.x * blockDim.x + threadIdx.x;
    if (row >= Mtot * 64) return;
    const int c = row >> 6, k = row & 63;
    float* fr = F + (size_t)row * Cpad;
    if (k < cnt[c]) {
        const int j = nbr[row];
        const float* xj = xsrc + (size_t)j * Cx;
        for (int t = 0; t < Cx; ++t) fr[t] = xj[t];
        const float* pj = psrc + (size_t)j * 3;
        const float* qc = q + (size_t)c * 3;
        fr[Cx+0] = pj[0] - qc[0];
        fr[Cx+1] = pj[1] - qc[1];
        fr[Cx+2] = pj[2] - qc[2];
        for (int t = Cx + 3; t < Cpad; ++t) fr[t] = 0.0f;
    } else {
        for (int t = 0; t < Cpad; ++t) fr[t] = 0.0f;
    }
}

// ---------------------------------------------------------------------------
// Copy weight [K,C] + bias [C] into zero-padded [Kp,Cp] + [Cp].
// ---------------------------------------------------------------------------
__global__ void pad_wb(const float* __restrict__ W, const float* __restrict__ b,
                       float* __restrict__ Wp, float* __restrict__ bp,
                       int K, int C, int Kp, int Cp)
{
    int i = blockIdx.x * blockDim.x + threadIdx.x;
    const int tot = Kp * Cp;
    if (i < tot) {
        const int r = i / Cp, c = i % Cp;
        Wp[i] = (r < K && c < C) ? W[(size_t)r * C + c] : 0.0f;
    } else if (i < tot + Cp) {
        const int c = i - tot;
        bp[c] = (c < C) ? b[c] : 0.0f;
    }
}

// ---------------------------------------------------------------------------
// GEMM: Y[M,Creal] = act(X[M,Kp]*W[Kp,Cp]+b), f16 WMMA, f32 accum.
// Kp % 32 == 0, Cp % 64 == 0 (padding guaranteed); only M-edge blocks guard.
// Block tile 64Mx64N, 8 waves, wave tile 16x32 (2 WMMAs share an A fragment).
// ---------------------------------------------------------------------------
__global__ __launch_bounds__(256) void gemm_bias_relu(
    const float* __restrict__ X, const float* __restrict__ W,
    const float* __restrict__ bias, float* __restrict__ Y,
    int M, int Kp, int Cp, int Creal, int ldy, int relu)
{
    __shared__ _Float16 As[TM][LDP];     // As[m][k]
    __shared__ _Float16 Bt[TN][LDP];     // Bt[n][k]  (transposed)

    const int tid  = threadIdx.x;
    const int lane = tid & 31;
    const int wave = tid >> 5;
    const int wm = wave & 3;             // 4 M-subtiles of 16
    const int wn = wave >> 2;            // 2 N-subtiles of 32
    const int tileM = blockIdx.y * TM;
    const int tileN = blockIdx.x * TN;
    const int hgrp = lane >> 4;
    const int l15  = lane & 15;

    v8f acc0 = {0.f,0.f,0.f,0.f,0.f,0.f,0.f,0.f};
    v8f acc1 = {0.f,0.f,0.f,0.f,0.f,0.f,0.f,0.f};

    const bool mfull = (tileM + TM <= M);

    for (int k0 = 0; k0 < Kp; k0 += TK) {
        if (mfull) {
#pragma unroll
            for (int t = 0; t < 2; ++t) {
                const int idx = tid + t * 256;
                const int r  = idx >> 3;            // 0..63
                const int c4 = (idx & 7) << 2;      // 0,4,..,28
                const float4 v = *(const float4*)&X[(size_t)(tileM + r) * Kp + k0 + c4];
                h4 qv = {(_Float16)v.x,(_Float16)v.y,(_Float16)v.z,(_Float16)v.w};
                *(h4*)&As[r][c4] = qv;
            }
        } else {
            for (int i = tid; i < TM * TK; i += 256) {
                const int r = i >> 5, c = i & 31;
                const int gm = tileM + r;
                As[r][c] = (_Float16)((gm < M) ? X[(size_t)gm * Kp + k0 + c] : 0.0f);
            }
        }
#pragma unroll
        for (int t = 0; t < 2; ++t) {
            const int idx = tid + t * 256;
            const int r  = idx >> 4;            // 0..31 (k)
            const int c4 = (idx & 15) << 2;     // 0,4,..,60 (n)
            const float4 v = *(const float4*)&W[(size_t)(k0 + r) * Cp + tileN + c4];
            Bt[c4+0][r] = (_Float16)v.x;
            Bt[c4+1][r] = (_Float16)v.y;
            Bt[c4+2][r] = (_Float16)v.z;
            Bt[c4+3][r] = (_Float16)v.w;
        }
        __syncthreads();

        const int mr  = wm * 16 + l15;
        const int nc0 = wn * 32 + l15;
        union Frag { uint4 u[2]; v16h h; };
        Frag fa, fb0, fb1;
        fa.u[0]  = *(const uint4*)&As[mr][16 * hgrp];
        fa.u[1]  = *(const uint4*)&As[mr][16 * hgrp + 8];
        fb0.u[0] = *(const uint4*)&Bt[nc0][16 * hgrp];
        fb0.u[1] = *(const uint4*)&Bt[nc0][16 * hgrp + 8];
        fb1.u[0] = *(const uint4*)&Bt[nc0 + 16][16 * hgrp];
        fb1.u[1] = *(const uint4*)&Bt[nc0 + 16][16 * hgrp + 8];

        acc0 = __builtin_amdgcn_wmma_f32_16x16x32_f16(
            false, fa.h, false, fb0.h, (short)0, acc0, false, false);
        acc1 = __builtin_amdgcn_wmma_f32_16x16x32_f16(
            false, fa.h, false, fb1.h, (short)0, acc1, false, false);
        __syncthreads();
    }

    const int nOut0 = tileN + wn * 32 + l15;
    const int nOut1 = nOut0 + 16;
    const float bv0 = (nOut0 < Cp) ? bias[nOut0] : 0.0f;
    const float bv1 = (nOut1 < Cp) ? bias[nOut1] : 0.0f;
#pragma unroll
    for (int r = 0; r < 8; ++r) {
        const int mOut = tileM + wm * 16 + r + 8 * hgrp;
        if (mOut < M) {
            if (nOut0 < Creal) {
                float v = acc0[r] + bv0;
                if (relu) v = v > 0.0f ? v : 0.0f;
                Y[(size_t)mOut * ldy + nOut0] = v;
            }
            if (nOut1 < Creal) {
                float v = acc1[r] + bv1;
                if (relu) v = v > 0.0f ? v : 0.0f;
                Y[(size_t)mOut * ldy + nOut1] = v;
            }
        }
    }
}

// ---------------------------------------------------------------------------
__global__ void maxpool_sa(const float* __restrict__ H, const int* __restrict__ cnt,
                           float* __restrict__ out, int Mtot, int C)
{
    int i = blockIdx.x * blockDim.x + threadIdx.x;
    if (i >= Mtot * C) return;
    const int c = i / C, ch = i % C;
    const int n = cnt[c];
    float v = -1e10f;
    const float* h = H + ((size_t)c * 64) * C + ch;
    for (int k = 0; k < n; ++k) v = fmaxf(v, h[(size_t)k * C]);
    out[i] = v;
}

// ---------------------------------------------------------------------------
__global__ void global_maxpool(const float* __restrict__ H, float* __restrict__ g,
                               int Bn, int Npts, int C)
{
    int i = blockIdx.x * blockDim.x + threadIdx.x;
    if (i >= Bn * C) return;
    const int b = i / C, ch = i % C;
    float v = -1e10f;
    const float* h = H + (size_t)b * Npts * C + ch;
    for (int k = 0; k < Npts; ++k) v = fmaxf(v, h[(size_t)k * C]);
    g[i] = v;
}

// ---------------------------------------------------------------------------
__global__ void concat2p(const float* __restrict__ A, int Ca,
                         const float* __restrict__ Bv, int Cb,
                         float* __restrict__ out, int M, int Cpad)
{
    int i = blockIdx.x * blockDim.x + threadIdx.x;
    if (i >= M * Cpad) return;
    const int r = i / Cpad, c = i % Cpad;
    float v = 0.0f;
    if (c < Ca)           v = A[(size_t)r * Ca + c];
    else if (c < Ca + Cb) v = Bv[(size_t)r * Cb + (c - Ca)];
    out[i] = v;
}

// ---------------------------------------------------------------------------
__global__ void fp0_build(const float* __restrict__ g, const float* __restrict__ x2,
                          float* __restrict__ F, int Bn, int N2_, int Cg, int Cx)
{
    const int Ct = Cg + Cx;
    int i = blockIdx.x * blockDim.x + threadIdx.x;
    if (i >= Bn * N2_ * Ct) return;
    const int r = i / Ct, c = i % Ct;
    const int b = r / N2_;
    F[i] = (c < Cg) ? g[(size_t)b * Cg + c] : x2[(size_t)r * Cx + (c - Cg)];
}

// ---------------------------------------------------------------------------
__global__ void knn3_interp(const float* __restrict__ fsrc,
                            const float* __restrict__ psrc,
                            const float* __restrict__ ptgt,
                            float* __restrict__ fout,
                            int Bn, int Ns, int Nt, int C)
{
    int t = blockIdx.x * blockDim.x + threadIdx.x;
    if (t >= Bn * Nt) return;
    const int b = t / Nt;
    const float* pt = ptgt + (size_t)t * 3;
    const float qx = pt[0], qy = pt[1], qz = pt[2];
    const float* pb = psrc + (size_t)b * Ns * 3;

    float d0 = 1e30f, d1 = 1e30f, d2 = 1e30f;
    int   i0 = 0,     i1 = 0,     i2 = 0;
    for (int j = 0; j < Ns; ++j) {
        const float dx = pb[j*3]-qx, dy = pb[j*3+1]-qy, dz = pb[j*3+2]-qz;
        const float d  = dx*dx + dy*dy + dz*dz;
        if (d < d0)      { d2=d1; i2=i1; d1=d0; i1=i0; d0=d; i0=j; }
        else if (d < d1) { d2=d1; i2=i1; d1=d;  i1=j; }
        else if (d < d2) { d2=d;  i2=j; }
    }
    const float w0 = 1.0f / fmaxf(d0, 1e-16f);
    const float w1 = 1.0f / fmaxf(d1, 1e-16f);
    const float w2 = 1.0f / fmaxf(d2, 1e-16f);
    const float wsum = w0 + w1 + w2;
    const float* f0 = fsrc + (size_t)(b * Ns + i0) * C;
    const float* f1 = fsrc + (size_t)(b * Ns + i1) * C;
    const float* f2 = fsrc + (size_t)(b * Ns + i2) * C;
    float* fo = fout + (size_t)t * C;
    for (int ch = 0; ch < C; ++ch)
        fo[ch] = (w0*f0[ch] + w1*f1[ch] + w2*f2[ch]) / wsum;
}

// ===========================================================================
extern "C" void kernel_launch(void* const* d_in, const int* in_sizes, int n_in,
                              void* d_out, int out_size, void* d_ws, size_t ws_size,
                              hipStream_t stream)
{
    (void)in_sizes; (void)n_in; (void)out_size; (void)ws_size;

    const int Bc = 16, N = 4096, N1 = 820, N2 = 205, FEAT = 4;
    const float R1 = 0.2f, R2 = 0.4f;

    const float* x   = (const float*)d_in[0];
    const float* pos = (const float*)d_in[1];

    char* wp = (char*)d_ws;
    size_t off = 0;
    auto alloc = [&](size_t bytes) -> char* {
        char* p = wp + off;
        off += (bytes + 255) & ~(size_t)255;
        return p;
    };
    float* bufA   = (float*)alloc((size_t)839680 * 64  * sizeof(float));
    float* bufB   = (float*)alloc((size_t)839680 * 128 * sizeof(float));
    float* fint   = (float*)alloc((size_t)65536  * 128 * sizeof(float));
    int*   idx1   = (int*)  alloc((size_t)Bc * N1 * sizeof(int));
    float* p1     = (float*)alloc((size_t)Bc * N1 * 3 * sizeof(float));
    int*   idx2   = (int*)  alloc((size_t)Bc * N2 * sizeof(int));
    float* p2     = (float*)alloc((size_t)Bc * N2 * 3 * sizeof(float));
    int*   nbr    = (int*)  alloc((size_t)Bc * N1 * 64 * sizeof(int));
    int*   cnt    = (int*)  alloc((size_t)Bc * N1 * sizeof(int));
    float* x1     = (float*)alloc((size_t)Bc * N1 * 128 * sizeof(float));
    float* x2     = (float*)alloc((size_t)Bc * N2 * 256 * sizeof(float));
    float* gfeat  = (float*)alloc((size_t)Bc * 1024 * sizeof(float));

    auto cdiv = [](long long a, long long b) -> int { return (int)((a + b - 1) / b); };

    // ---- padded weights (K -> mult of 32, C -> mult of 64) in workspace ----
    const int NL = 19;
    const int lK [NL] = {  7, 64, 64,131,128,128,259,256, 512,1280,256,384,256,132,128,128,128,128,128};
    const int lC [NL] = { 64, 64,128,128,128,256,256,512,1024, 256,256,256,128,128,128,128,128,128, 16};
    const int lKp[NL] = { 32, 64, 64,160,128,128,288,256, 512,1280,256,384,256,160,128,128,128,128,128};
    const int lCp[NL] = { 64, 64,128,128,128,256,256,512,1024, 256,256,256,128,128,128,128,128,128, 64};
    float* Wp[NL];
    float* bp[NL];
    for (int l = 0; l < NL; ++l) {
        Wp[l] = (float*)alloc((size_t)lKp[l] * lCp[l] * sizeof(float));
        bp[l] = (float*)alloc((size_t)lCp[l] * sizeof(float));
        const float* Wsrc = (const float*)d_in[3 + 2*l];
        const float* bsrc = (const float*)d_in[4 + 2*l];
        const int tot = lKp[l] * lCp[l] + lCp[l];
        pad_wb<<<cdiv(tot, 256), 256, 0, stream>>>(Wsrc, bsrc, Wp[l], bp[l],
                                                   lK[l], lC[l], lKp[l], lCp[l]);
    }

    auto gemm = [&](const float* X_, int layer, float* Y_, int M_, int Creal, int ldy, int relu_) {
        dim3 g(cdiv(lCp[layer], TN), cdiv(M_, TM));
        gemm_bias_relu<<<g, 256, 0, stream>>>(X_, Wp[layer], bp[layer], Y_,
                                              M_, lKp[layer], lCp[layer], Creal, ldy, relu_);
    };

    // ---------------- SA level 1 ----------------
    fps_kernel<<<Bc, 256, 0, stream>>>(pos, N, N1, idx1);
    gather3<<<cdiv((long long)Bc*N1*3, 256), 256, 0, stream>>>(pos, idx1, p1, Bc*N1);
    ball_query<<<cdiv((long long)Bc*N1, 8), 256, 0, stream>>>(pos, p1, R1*R1, nbr, cnt, Bc, N, N1);
    {
        const int Mtot = Bc * N1;                     // 13120 centers, 839680 rows
        build_sa_feat<<<cdiv((long long)Mtot*64, 256), 256, 0, stream>>>(x, FEAT, pos, p1, nbr, cnt, bufA, Mtot, 32);
        gemm(bufA, 0, bufB, Mtot*64, 64,  64,  1);
        gemm(bufB, 1, bufA, Mtot*64, 64,  64,  1);
        gemm(bufA, 2, bufB, Mtot*64, 128, 128, 1);
        maxpool_sa<<<cdiv((long long)Mtot*128, 256), 256, 0, stream>>>(bufB, cnt, x1, Mtot, 128);
    }

    // ---------------- SA level 2 ----------------
    fps_kernel<<<Bc, 256, 0, stream>>>(p1, N1, N2, idx2);
    gather3<<<cdiv((long long)Bc*N2*3, 256), 256, 0, stream>>>(p1, idx2, p2, Bc*N2);
    ball_query<<<cdiv((long long)Bc*N2, 8), 256, 0, stream>>>(p1, p2, R2*R2, nbr, cnt, Bc, N1, N2);
    {
        const int Mtot = Bc * N2;                     // 3280 centers, 209920 rows
        build_sa_feat<<<cdiv((long long)Mtot*64, 256), 256, 0, stream>>>(x1, 128, p1, p2, nbr, cnt, bufA, Mtot, 160);
        gemm(bufA, 3, bufB, Mtot*64, 128, 128, 1);
        gemm(bufB, 4, bufA, Mtot*64, 128, 128, 1);
        gemm(bufA, 5, bufB, Mtot*64, 256, 256, 1);
        maxpool_sa<<<cdiv((long long)Mtot*256, 256), 256, 0, stream>>>(bufB, cnt, x2, Mtot, 256);
    }

    // ---------------- Global SA ----------------
    {
        const int Mtot = Bc * N2;
        concat2p<<<cdiv((long long)Mtot*288, 256), 256, 0, stream>>>(x2, 256, p2, 3, bufA, Mtot, 288);
        gemm(bufA, 6, bufB, Mtot, 256,  256,  1);
        gemm(bufB, 7, bufA, Mtot, 512,  512,  1);
        gemm(bufA, 8, bufB, Mtot, 1024, 1024, 1);
        global_maxpool<<<cdiv((long long)Bc*1024, 256), 256, 0, stream>>>(bufB, gfeat, Bc, N2, 1024);
    }

    // ---------------- FP0 (broadcast global -> level-2 points) ----------------
    {
        const int Mtot = Bc * N2;
        fp0_build<<<cdiv((long long)Mtot*1280, 256), 256, 0, stream>>>(gfeat, x2, bufA, Bc, N2, 1024, 256);
        gemm(bufA, 9,  bufB, Mtot, 256, 256, 1);
        gemm(bufB, 10, bufA, Mtot, 256, 256, 1);      // f@L2 in bufA
    }

    // ---------------- FP1 (L2 -> L1) ----------------
    {
        knn3_interp<<<cdiv((long long)Bc*N1, 256), 256, 0, stream>>>(bufA, p2, p1, fint, Bc, N2, N1, 256);
        const int Mtot = Bc * N1;
        concat2p<<<cdiv((long long)Mtot*384, 256), 256, 0, stream>>>(fint, 256, x1, 128, bufB, Mtot, 384);
        gemm(bufB, 11, bufA, Mtot, 256, 256, 1);
        gemm(bufA, 12, bufB, Mtot, 128, 128, 1);      // f@L1 in bufB
    }

    // ---------------- FP2 (L1 -> original) + head ----------------
    {
        knn3_interp<<<cdiv((long long)Bc*N, 256), 256, 0, stream>>>(bufB, p1, pos, fint, Bc, N1, N, 128);
        const int Mtot = Bc * N;
        concat2p<<<cdiv((long long)Mtot*160, 256), 256, 0, stream>>>(fint, 128, x, FEAT, bufA, Mtot, 160);
        gemm(bufA, 13, bufB, Mtot, 128, 128, 1);
        gemm(bufB, 14, bufA, Mtot, 128, 128, 1);
        gemm(bufA, 15, bufB, Mtot, 128, 128, 1);
        // lin head (ReLU both layers)
        gemm(bufB, 16, bufA, Mtot, 128, 128, 1);
        gemm(bufA, 17, bufB, Mtot, 128, 128, 1);
        // output projection, no activation: store straight into d_out (pitch 16)
        gemm(bufB, 18, (float*)d_out, Mtot, 16, 16, 0);
    }
}
